// GVPGraphEmbedding_42417097015953
// MI455X (gfx1250) — compile-verified, matrix operand order checked by reading, not run on hardware
//
#include <hip/hip_runtime.h>
#include <math.h>

typedef _Float16 v16h __attribute__((ext_vector_type(16)));
typedef _Float16 v8h  __attribute__((ext_vector_type(8)));
typedef float    v8f  __attribute__((ext_vector_type(8)));

#define NB 8
#define NL 2048
#define NTOPK 30
#define NNODES (NB*NL)          // 16384
#define EPB (NL*NTOPK)          // 61440 edges per batch
#define NEDGES (NNODES*NTOPK)   // 491520
#define DSZ 1024
#define DV 256
#define ESD 32
#define KN 263                  // node scalar feature count (7 + 256)
#define KNP 288                 // padded to multiple of 32
#define KE 64                   // edge feature K padded (35 -> 64)

// ---------- small vector helpers ----------
struct F3 { float x, y, z; };
__device__ __forceinline__ F3 sub3(F3 a, F3 b){ return {a.x-b.x, a.y-b.y, a.z-b.z}; }
__device__ __forceinline__ F3 add3(F3 a, F3 b){ return {a.x+b.x, a.y+b.y, a.z+b.z}; }
__device__ __forceinline__ F3 scl3(F3 a, float s){ return {a.x*s, a.y*s, a.z*s}; }
__device__ __forceinline__ float dot3(F3 a, F3 b){ return a.x*b.x + a.y*b.y + a.z*b.z; }
__device__ __forceinline__ F3 crs3(F3 a, F3 b){
  return {a.y*b.z - a.z*b.y, a.z*b.x - a.x*b.z, a.x*b.y - a.y*b.x};
}
__device__ __forceinline__ F3 nrm3(F3 a){
  float n = sqrtf(dot3(a,a) + 1e-8f);
  return scl3(a, 1.0f/n);
}
__device__ __forceinline__ F3 ldp3(const float* p){ return {p[0], p[1], p[2]}; }

// ---------- init: zero the per-batch E_vec accumulators ----------
__global__ void init_sums_kernel(float* sums){
  int t = threadIdx.x;
  if (t < 32) sums[t] = 0.0f;
}

// ---------- geometry: node_s (7) and node_v (3x3) per node ----------
__global__ __launch_bounds__(256)
void geom_kernel(const float* __restrict__ coords,
                 const unsigned char* __restrict__ cmask,
                 float* __restrict__ gS, float* __restrict__ gV)
{
  int idx = blockIdx.x * blockDim.x + threadIdx.x;
  if (idx >= NNODES) return;
  int b = idx / NL, l = idx % NL;
  const float* Xb = coords + (size_t)b * NL * 9;   // [L][3 atoms][3]

  float Dv[3];
  #pragma unroll
  for (int c = 0; c < 3; ++c){
    int m = 3*l + c - 1;
    if (m < 0 || m > 3*NL - 4){ Dv[c] = 0.0f; continue; }
    F3 p0 = ldp3(Xb + (size_t)m*3);
    F3 p1 = ldp3(Xb + (size_t)(m+1)*3);
    F3 p2 = ldp3(Xb + (size_t)(m+2)*3);
    F3 p3 = ldp3(Xb + (size_t)(m+3)*3);
    F3 u2 = nrm3(sub3(p1, p0));
    F3 u1 = nrm3(sub3(p2, p1));
    F3 u0 = nrm3(sub3(p3, p2));
    F3 n2 = nrm3(crs3(u2, u1));
    F3 n1 = nrm3(crs3(u1, u0));
    float cd = dot3(n2, n1);
    cd = fminf(fmaxf(cd, -1.0f + 1e-7f), 1.0f - 1e-7f);
    float sg = dot3(u2, n1);
    float s  = (sg > 0.0f) ? 1.0f : ((sg < 0.0f) ? -1.0f : 0.0f);
    Dv[c] = s * acosf(cd);
  }
  float* So = gS + (size_t)idx * 7;
  So[0] = cosf(Dv[0]); So[1] = cosf(Dv[1]); So[2] = cosf(Dv[2]);
  So[3] = sinf(Dv[0]); So[4] = sinf(Dv[1]); So[5] = sinf(Dv[2]);
  So[6] = cmask[idx] ? 1.0f : 0.0f;

  F3 xc = ldp3(Xb + (size_t)(l*3 + 1)*3);
  F3 fwd = {0,0,0}, bwd = {0,0,0};
  if (l < NL-1) fwd = nrm3(sub3(ldp3(Xb + (size_t)((l+1)*3 + 1)*3), xc));
  if (l > 0)    bwd = nrm3(sub3(ldp3(Xb + (size_t)((l-1)*3 + 1)*3), xc));

  F3 natm = ldp3(Xb + (size_t)(l*3 + 0)*3);
  F3 oatm = ldp3(Xb + (size_t)(l*3 + 1)*3);
  F3 catm = ldp3(Xb + (size_t)(l*3 + 2)*3);
  F3 cv = nrm3(sub3(catm, oatm));
  F3 nv = nrm3(sub3(natm, oatm));
  F3 bis = nrm3(add3(cv, nv));
  F3 pp  = nrm3(crs3(cv, nv));
  const float a = 0.5773502691896258f, bb = 0.816496580927726f;
  F3 sd = { -bis.x*a - pp.x*bb, -bis.y*a - pp.y*bb, -bis.z*a - pp.z*bb };

  float* Vo = gV + (size_t)idx * 9;
  Vo[0]=fwd.x; Vo[1]=fwd.y; Vo[2]=fwd.z;
  Vo[3]=bwd.x; Vo[4]=bwd.y; Vo[5]=bwd.z;
  Vo[6]=sd.x;  Vo[7]=sd.y;  Vo[8]=sd.z;
}

// ---------- node prep: vh = v^T @ wh (3x3 -> 3x256), vn, f16 feature row ----------
__global__ __launch_bounds__(256)
void node_prep_kernel(const float* __restrict__ gS, const float* __restrict__ gV,
                      const float* __restrict__ wh,
                      _Float16* __restrict__ feat, _Float16* __restrict__ vhA)
{
  int node = blockIdx.x;
  int h = threadIdx.x;                 // 0..255 hidden channel
  const float* V = gV + (size_t)node * 9;   // [vec][xyz]
  float w0 = wh[h], w1 = wh[256 + h], w2 = wh[512 + h];
  float vx = V[0]*w0 + V[3]*w1 + V[6]*w2;
  float vy = V[1]*w0 + V[4]*w1 + V[7]*w2;
  float vz = V[2]*w0 + V[5]*w1 + V[8]*w2;
  float vn = sqrtf(fmaxf(vx*vx + vy*vy + vz*vz, 1e-8f));

  _Float16* frow = feat + (size_t)node * KNP;
  frow[7 + h] = (_Float16)vn;
  if (h < 7)           frow[h]       = (_Float16)gS[(size_t)node*7 + h];
  if (h < (KNP - KN))  frow[KN + h]  = (_Float16)0.0f;

  _Float16* vr = vhA + (size_t)node * 3 * DV;   // rows: node*3 + xyz, ld = 256
  vr[h]        = (_Float16)vx;
  vr[DV + h]   = (_Float16)vy;
  vr[2*DV + h] = (_Float16)vz;
}

// ---------- weight packing into WMMA B-fragment order ----------
// Packed layout: [(kt*Ntiles + nt)][lane 0..31][16 f16], matching the f16
// 16x16x32 per-lane K pattern (vgpr0-3 -> k {half*8..}, vgpr4-7 -> k {16+half*8..}).
__global__ void pack_kernel(const float* __restrict__ W, int Krows, int Ncols,
                            int Ksteps, int Ntiles, _Float16* __restrict__ out)
{
  int idx = blockIdx.x * blockDim.x + threadIdx.x;
  int total = Ksteps * Ntiles * 512;
  if (idx >= total) return;
  int i16  = idx & 15;
  int lane = (idx >> 4) & 31;
  int tile = idx >> 9;
  int nt = tile % Ntiles;
  int kt = tile / Ntiles;
  int half = lane >> 4;
  int vg = i16 >> 1, ppos = i16 & 1;
  int kl = (vg < 4) ? (vg*2 + ppos + half*8) : (16 + (vg-4)*2 + ppos + half*8);
  int k = kt*32 + kl;
  int n = nt*16 + (lane & 15);
  float val = (k < Krows && n < Ncols) ? W[(size_t)k * Ncols + n] : 0.0f;
  out[idx] = (_Float16)val;
}

// ---------- top-k: one block per (b,i) row; candidates in registers, LDS argmin ----------
__global__ __launch_bounds__(256)
void topk_kernel(const float* __restrict__ coords,
                 const unsigned char* __restrict__ cmask,
                 const unsigned char* __restrict__ pmask,
                 int* __restrict__ Eidx, float* __restrict__ Edist)
{
  __shared__ float sval[256];
  __shared__ int   sidx[256];
  __shared__ int   schosen;
  int bid = blockIdx.x;
  int b = bid / NL, i = bid % NL;
  int t = threadIdx.x;
  const float* Xb = coords + (size_t)b * NL * 9;
  float xix = Xb[(size_t)(i*3+1)*3 + 0];
  float xiy = Xb[(size_t)(i*3+1)*3 + 1];
  float xiz = Xb[(size_t)(i*3+1)*3 + 2];
  int cmi = cmask[(size_t)b*NL + i] ? 1 : 0;
  int rmi = pmask[(size_t)b*NL + i] ? 0 : 1;

  float vals[8];
  #pragma unroll
  for (int q = 0; q < 8; ++q){
    int j = q*256 + t;
    float dx = xix - Xb[(size_t)(j*3+1)*3 + 0];
    float dy = xiy - Xb[(size_t)(j*3+1)*3 + 1];
    float dz = xiz - Xb[(size_t)(j*3+1)*3 + 2];
    float dist = sqrtf(dx*dx + dy*dy + dz*dz + 1e-8f);
    int cmj = cmask[(size_t)b*NL + j] ? 1 : 0;
    int rmj = pmask[(size_t)b*NL + j] ? 0 : 1;
    int cm2 = cmi & cmj;
    int rm2 = rmi & rmj;
    float d = cm2 ? dist : 0.0f;
    if (!cm2) d += 1e8f + fabsf((float)(i - j)) * 1e6f;
    if (!rm2) d += 1e10f;
    vals[q] = d;
  }

  unsigned sel = 0u;
  for (int r = 0; r < NTOPK; ++r){
    float mv = 3.4e38f; int mi = 1 << 30;
    #pragma unroll
    for (int q = 0; q < 8; ++q){
      if (sel & (1u << q)) continue;
      int j = q*256 + t;
      float v = vals[q];
      if (v < mv || (v == mv && j < mi)){ mv = v; mi = j; }
    }
    sval[t] = mv; sidx[t] = mi;
    __syncthreads();
    for (int s = 128; s > 0; s >>= 1){
      if (t < s){
        float ov = sval[t+s]; int oi = sidx[t+s];
        if (ov < sval[t] || (ov == sval[t] && oi < sidx[t])){ sval[t] = ov; sidx[t] = oi; }
      }
      __syncthreads();
    }
    if (t == 0){
      Edist[(size_t)bid*NTOPK + r] = sval[0];
      Eidx [(size_t)bid*NTOPK + r] = sidx[0];
      schosen = sidx[0];
    }
    __syncthreads();
    int ch = schosen;
    if ((ch & 255) == t) sel |= 1u << (ch >> 8);
    __syncthreads();
  }
}

// ---------- edge pass A: E_vec, per-batch masked sums ----------
__global__ __launch_bounds__(256)
void edgeA_kernel(const float* __restrict__ coords,
                  const int* __restrict__ Eidx, const float* __restrict__ Edist,
                  float4* __restrict__ evec, float* __restrict__ sums)
{
  __shared__ float rx[256], ry[256], rz[256], rw[256];
  int b  = blockIdx.x / 240;
  int eb = (blockIdx.x % 240) * 256 + threadIdx.x;   // 0..61439
  int e  = b * EPB + eb;
  int i  = eb / NTOPK;
  int j  = Eidx[e];
  float dist = Edist[e];
  const float* Xb = coords + (size_t)b * NL * 9;
  float vx = Xb[(size_t)(i*3+1)*3+0] - Xb[(size_t)(j*3+1)*3+0];
  float vy = Xb[(size_t)(i*3+1)*3+1] - Xb[(size_t)(j*3+1)*3+1];
  float vz = Xb[(size_t)(i*3+1)*3+2] - Xb[(size_t)(j*3+1)*3+2];
  float w = (dist < 5e7f) ? 1.0f : 0.0f;
  evec[e] = make_float4(vx, vy, vz, w);

  int t = threadIdx.x;
  rx[t] = vx*w; ry[t] = vy*w; rz[t] = vz*w; rw[t] = w;
  __syncthreads();
  for (int s = 128; s > 0; s >>= 1){
    if (t < s){ rx[t]+=rx[t+s]; ry[t]+=ry[t+s]; rz[t]+=rz[t+s]; rw[t]+=rw[t+s]; }
    __syncthreads();
  }
  if (t == 0){
    atomicAdd(&sums[b*4+0], rx[0]);
    atomicAdd(&sums[b*4+1], ry[0]);
    atomicAdd(&sums[b*4+2], rz[0]);
    atomicAdd(&sums[b*4+3], rw[0]);
  }
}

// ---------- edge pass B: edge feature rows (f16), ev output, ei output ----------
__global__ __launch_bounds__(256)
void edgeB_kernel(const int* __restrict__ Eidx, const float* __restrict__ Edist,
                  const float4* __restrict__ evec, const float* __restrict__ sums,
                  const unsigned char* __restrict__ cmask,
                  const float* __restrict__ edge_wh, const float* __restrict__ edge_wv,
                  _Float16* __restrict__ efeat, float* __restrict__ outEV,
                  int* __restrict__ outEI)
{
  int e = blockIdx.x * blockDim.x + threadIdx.x;
  if (e >= NEDGES) return;
  int b  = e / EPB;
  int eb = e % EPB;
  int i  = eb / NTOPK;
  int j  = Eidx[e];
  float dist = Edist[e];
  float4 f = evec[e];
  float sw = sums[b*4+3];
  float mx = sums[b*4+0] / sw, my = sums[b*4+1] / sw, mz = sums[b*4+2] / sw;
  float vx = (f.w > 0.0f) ? f.x : mx;
  float vy = (f.w > 0.0f) ? f.y : my;
  float vz = (f.w > 0.0f) ? f.z : mz;
  float nr = sqrtf(vx*vx + vy*vy + vz*vz + 1e-8f);
  float nx = vx/nr, ny = vy/nr, nz_ = vz/nr;
  float wh = edge_wh[0];
  float vhx = nx*wh, vhy = ny*wh, vhz = nz_*wh;
  float vn = sqrtf(fmaxf(vhx*vhx + vhy*vhy + vhz*vhz, 1e-8f));
  float wv = edge_wv[0];
  float ox = vhx*wv, oy = vhy*wv, oz = vhz*wv;
  float vnsq = fmaxf(ox*ox + oy*oy + oz*oz, 1e-4f);
  float nzf  = (vnsq > 2e-4f) ? 1.0f : 0.0f;
  float vnm  = (vnsq * nzf) / (1e-4f + nzf);
  float sc   = (nzf > 0.0f) ? (nzf * rsqrtf(vnm + 1e-4f)) : 0.0f;
  outEV[(size_t)e*3 + 0] = ox * sc;
  outEV[(size_t)e*3 + 1] = oy * sc;
  outEV[(size_t)e*3 + 2] = oz * sc;

  _Float16* fr = efeat + (size_t)e * KE;
  #pragma unroll
  for (int m = 0; m < 16; ++m){
    float z = (dist - (float)m * (20.0f/15.0f)) * (1.0f/1.25f);
    fr[m] = (_Float16)expf(-z*z);
  }
  float d = (float)(i - j);
  #pragma unroll
  for (int m = 0; m < 8; ++m){
    float fq = expf(-(float)(2*m) * (9.210340371976184f / 16.0f));
    float ang = d * fq;
    fr[16 + m] = (_Float16)cosf(ang);
    fr[24 + m] = (_Float16)sinf(ang);
  }
  fr[32] = (_Float16)(cmask[(size_t)b*NL + i] ? 0.0f : 1.0f);
  fr[33] = (_Float16)(cmask[(size_t)b*NL + j] ? 0.0f : 1.0f);
  fr[34] = (_Float16)vn;
  #pragma unroll
  for (int m = 35; m < KE; ++m) fr[m] = (_Float16)0.0f;

  int valid = (dist < 5e7f) ? 1 : 0;
  outEI[e]          = valid ? (i + b*NL) : -1;
  outEI[NEDGES + e] = valid ? (j + b*NL) : -1;
}

// ---------- WMMA GEMM with async-LDS staged B panels ----------
// Block = 8 waves: 8 M-tiles x NTB N-tiles. B panels for the current k-step are
// copied global->LDS with GLOBAL_LOAD_ASYNC_TO_LDS_B128 (ASYNCcnt), double
// buffered; each wave keeps one A fragment in registers and reuses it for NTB
// WMMAs. mode 0: row-major f32 out (+bias); mode 1: hv scatter [node][h][xyz].
template<int NTB>
__global__ __launch_bounds__(256)
void wmma_gemm_lds_kernel(const _Float16* __restrict__ A,
                          const _Float16* __restrict__ Bp,
                          const float* __restrict__ bias,
                          float* __restrict__ Out,
                          int Ntiles, int Ksteps, int ldA, int ldOut, int mode)
{
  __shared__ _Float16 smem[2 * NTB * 512];
  const int tid  = threadIdx.x;
  const int w    = tid >> 5;
  const int lane = tid & 31;
  const int half = lane >> 4;
  const int mr   = lane & 15;
  const int nBlocks = Ntiles / NTB;
  const int bm = blockIdx.x / nBlocks;
  const int bn = blockIdx.x % nBlocks;
  const int mt = bm * 8 + w;
  const int ntBase = bn * NTB;
  const int chunks = NTB * 64;      // 16-byte chunks per staged panel group

  union V16 { v16h v; v8h h[2]; };
  v8f zero = {};
  v8f acc[NTB];
  #pragma unroll
  for (int n = 0; n < NTB; ++n) acc[n] = zero;

  const _Float16* arow = A + (size_t)(mt*16 + mr) * ldA + half*8;

  auto stage = [&](int buf, int kt){
    if (tid < chunks){   // wave-uniform predicate (chunks is a multiple of 32)
      const _Float16* gsrc = Bp + (size_t)(kt*Ntiles + ntBase) * 512 + (size_t)tid * 8;
      unsigned lo = (unsigned)(size_t)(&smem[buf*NTB*512]) + (unsigned)(tid*16);
      unsigned long long ga = (unsigned long long)(size_t)gsrc;
      asm volatile("global_load_async_to_lds_b128 %0, %1, off"
                   :: "v"(lo), "v"(ga) : "memory");
    }
  };

  stage(0, 0);
  for (int kt = 0; kt < Ksteps; ++kt){
    int cur = kt & 1;
    if (kt + 1 < Ksteps){
      stage(cur ^ 1, kt + 1);
      asm volatile("s_wait_asynccnt 0x1" ::: "memory");   // kt's copy done (in-order)
    } else {
      asm volatile("s_wait_asynccnt 0x0" ::: "memory");
    }
    __syncthreads();

    V16 a;
    a.h[0] = *(const v8h*)(arow + kt*32);
    a.h[1] = *(const v8h*)(arow + kt*32 + 16);
    #pragma unroll
    for (int n = 0; n < NTB; ++n){
      V16 bf;
      const _Float16* bp = &smem[cur*NTB*512 + (n*32 + lane)*16];
      bf.h[0] = *(const v8h*)(bp);
      bf.h[1] = *(const v8h*)(bp + 8);
      acc[n] = __builtin_amdgcn_wmma_f32_16x16x32_f16(false, a.v, false, bf.v,
                                                      (short)0, acc[n], false, false);
    }
    __syncthreads();   // all reads of 'cur' done before it is restaged
  }

  int rbase = mt*16 + half*8;
  #pragma unroll
  for (int n = 0; n < NTB; ++n){
    int col = (ntBase + n)*16 + mr;
    if (mode == 0){
      float bv = bias ? bias[col] : 0.0f;
      #pragma unroll
      for (int r = 0; r < 8; ++r)
        Out[(size_t)(rbase + r) * ldOut + col] = acc[n][r] + bv;
    } else {
      #pragma unroll
      for (int r = 0; r < 8; ++r){
        int M = rbase + r;
        int node = M / 3, xyz = M - node*3;
        Out[(size_t)node * (DV*3) + col*3 + xyz] = acc[n][r];
      }
    }
  }
}

// ---------- hs layernorm (+ confidence RBF projection added post-LN) ----------
__global__ __launch_bounds__(256)
void ln_hs_kernel(float* __restrict__ hs, const float* __restrict__ g,
                  const float* __restrict__ bn, const float* __restrict__ confidence,
                  const float* __restrict__ conf_w, const float* __restrict__ conf_b)
{
  __shared__ float red[256];
  __shared__ float s_mu, s_rstd;
  int node = blockIdx.x;
  int t = threadIdx.x;
  float* row = hs + (size_t)node * DSZ;

  float x[4]; float sum = 0.0f;
  #pragma unroll
  for (int u = 0; u < 4; ++u){ x[u] = row[u*256 + t]; sum += x[u]; }
  red[t] = sum; __syncthreads();
  for (int s = 128; s > 0; s >>= 1){ if (t < s) red[t] += red[t+s]; __syncthreads(); }
  if (t == 0) s_mu = red[0] * (1.0f/DSZ);
  __syncthreads();
  float mu = s_mu;
  float vs = 0.0f;
  #pragma unroll
  for (int u = 0; u < 4; ++u){ float dd = x[u] - mu; vs += dd*dd; }
  red[t] = vs; __syncthreads();
  for (int s = 128; s > 0; s >>= 1){ if (t < s) red[t] += red[t+s]; __syncthreads(); }
  if (t == 0) s_rstd = rsqrtf(red[0] * (1.0f/DSZ) + 1e-4f);
  __syncthreads();
  float rstd = s_rstd;

  float conf = confidence[node];
  float r16[16];
  #pragma unroll
  for (int m = 0; m < 16; ++m){
    float z = (conf - (float)m * (1.0f/15.0f)) * 16.0f;
    r16[m] = expf(-z*z);
  }
  #pragma unroll
  for (int u = 0; u < 4; ++u){
    int c = u*256 + t;
    float cc = 0.0f;
    #pragma unroll
    for (int m = 0; m < 16; ++m) cc += r16[m] * conf_w[(size_t)m*DSZ + c];
    row[c] = (x[u] - mu) * rstd * g[c] + bn[c] + cc + conf_b[c];
  }
}

// ---------- hv vector layernorm (per node over 256 vectors) ----------
__global__ __launch_bounds__(256)
void ln_hv_kernel(float* __restrict__ hv)
{
  __shared__ float r1[256], r2[256];
  int node = blockIdx.x;
  int t = threadIdx.x;
  float* p = hv + (size_t)node * (DV*3) + t*3;
  float x = p[0], y = p[1], z = p[2];
  float vn = fmaxf(x*x + y*y + z*z, 1e-4f);
  float nz = (vn > 2e-4f) ? 1.0f : 0.0f;
  r1[t] = vn * nz; r2[t] = nz;
  __syncthreads();
  for (int s = 128; s > 0; s >>= 1){
    if (t < s){ r1[t] += r1[t+s]; r2[t] += r2[t+s]; }
    __syncthreads();
  }
  float vnm = r1[0] / (1e-4f + r2[0]);
  float sc = nz * rsqrtf(vnm + 1e-4f);
  p[0] = x * sc; p[1] = y * sc; p[2] = z * sc;
}

// ---------- edge scalar layernorm (32 channels, one thread per edge) ----------
__global__ __launch_bounds__(256)
void ln_es_kernel(float* __restrict__ es, const float* __restrict__ g,
                  const float* __restrict__ bn)
{
  int e = blockIdx.x * blockDim.x + threadIdx.x;
  if (e >= NEDGES) return;
  float* row = es + (size_t)e * ESD;
  float v[ESD]; float mu = 0.0f;
  #pragma unroll
  for (int c = 0; c < ESD; ++c){ v[c] = row[c]; mu += v[c]; }
  mu *= (1.0f/ESD);
  float var = 0.0f;
  #pragma unroll
  for (int c = 0; c < ESD; ++c){ float d = v[c] - mu; var += d*d; }
  var *= (1.0f/ESD);
  float rstd = rsqrtf(var + 1e-4f);
  #pragma unroll
  for (int c = 0; c < ESD; ++c) row[c] = (v[c] - mu) * rstd * g[c] + bn[c];
}

// ---------- host-side launch ----------
extern "C" void kernel_launch(void* const* d_in, const int* in_sizes, int n_in,
                              void* d_out, int out_size, void* d_ws, size_t ws_size,
                              hipStream_t stream)
{
  (void)in_sizes; (void)n_in; (void)out_size; (void)ws_size;
  const float*         coords     = (const float*)d_in[0];
  const unsigned char* cmask      = (const unsigned char*)d_in[1];
  const unsigned char* pmask      = (const unsigned char*)d_in[2];
  const float*         confidence = (const float*)d_in[3];
  const float*         node_wh    = (const float*)d_in[4];
  const float*         node_ws_w  = (const float*)d_in[5];
  const float*         node_ws_b  = (const float*)d_in[6];
  const float*         node_wv    = (const float*)d_in[7];
  const float*         node_ln_g  = (const float*)d_in[8];
  const float*         node_ln_b  = (const float*)d_in[9];
  const float*         edge_wh    = (const float*)d_in[10];
  const float*         edge_ws_w  = (const float*)d_in[11];
  const float*         edge_ws_b  = (const float*)d_in[12];
  const float*         edge_wv    = (const float*)d_in[13];
  const float*         edge_ln_g  = (const float*)d_in[14];
  const float*         edge_ln_b  = (const float*)d_in[15];
  const float*         conf_w     = (const float*)d_in[16];
  const float*         conf_b     = (const float*)d_in[17];

  // output regions
  float* out = (float*)d_out;
  const size_t OFF_HS = 0;
  const size_t OFF_HV = OFF_HS + (size_t)NNODES * DSZ;      // 16,777,216
  const size_t OFF_ES = OFF_HV + (size_t)NNODES * DV * 3;   // 29,360,128
  const size_t OFF_EV = OFF_ES + (size_t)NEDGES * ESD;      // 45,088,768
  const size_t OFF_EI = OFF_EV + (size_t)NEDGES * 3;        // 46,563,328
  float* outHS = out + OFF_HS;
  float* outHV = out + OFF_HV;
  float* outES = out + OFF_ES;
  float* outEV = out + OFF_EV;
  int*   outEI = (int*)(out + OFF_EI);

  // workspace layout (256B aligned)
  char* ws = (char*)d_ws;
  size_t off = 0;
  auto carve = [&](size_t bytes)->char*{
    char* p = ws + off;
    off += (bytes + 255) & ~(size_t)255;
    return p;
  };
  float*     sums  = (float*)    carve(32 * sizeof(float));
  float*     gS    = (float*)    carve((size_t)NNODES * 7 * sizeof(float));
  float*     gV    = (float*)    carve((size_t)NNODES * 9 * sizeof(float));
  _Float16*  feat  = (_Float16*) carve((size_t)NNODES * KNP * sizeof(_Float16));
  _Float16*  vhA   = (_Float16*) carve((size_t)NNODES * 3 * DV * sizeof(_Float16));
  int*       Eidx  = (int*)      carve((size_t)NEDGES * sizeof(int));
  float*     Edist = (float*)    carve((size_t)NEDGES * sizeof(float));
  float4*    evec  = (float4*)   carve((size_t)NEDGES * sizeof(float4));
  _Float16*  efeat = (_Float16*) carve((size_t)NEDGES * KE * sizeof(_Float16));
  _Float16*  pWn   = (_Float16*) carve((size_t)9  * 64 * 512 * sizeof(_Float16));
  _Float16*  pWv   = (_Float16*) carve((size_t)8  * 16 * 512 * sizeof(_Float16));
  _Float16*  pWe   = (_Float16*) carve((size_t)2  * 2  * 512 * sizeof(_Float16));

  // 0) init accumulators
  init_sums_kernel<<<1, 64, 0, stream>>>(sums);

  // 1) weight packing into WMMA fragment order
  pack_kernel<<<(9*64*512 + 255)/256, 256, 0, stream>>>(node_ws_w, KN, DSZ, 9, 64, pWn);
  pack_kernel<<<(8*16*512 + 255)/256, 256, 0, stream>>>(node_wv, DV, DV, 8, 16, pWv);
  pack_kernel<<<(2*2*512  + 255)/256, 256, 0, stream>>>(edge_ws_w, 35, ESD, 2, 2, pWe);

  // 2) node geometry + feature assembly
  geom_kernel<<<NNODES/256, 256, 0, stream>>>(coords, cmask, gS, gV);
  node_prep_kernel<<<NNODES, 256, 0, stream>>>(gS, gV, node_wh, feat, vhA);

  // 3) top-k neighbors
  topk_kernel<<<NNODES, 256, 0, stream>>>(coords, cmask, pmask, Eidx, Edist);

  // 4) edge passes
  edgeA_kernel<<<NB*240, 256, 0, stream>>>(coords, Eidx, Edist, evec, sums);
  edgeB_kernel<<<NEDGES/256, 256, 0, stream>>>(Eidx, Edist, evec, sums, cmask,
                                               edge_wh, edge_wv, efeat, outEV, outEI);

  // 5) WMMA GEMMs (async-LDS staged B, A-register reuse)
  // hs: [16384,288] x [288,1024]  -> blocks of 128M x 64N
  wmma_gemm_lds_kernel<4><<<(1024/8)*(64/4), 256, 0, stream>>>(
      feat, pWn, node_ws_b, outHS, 64, 9, KNP, DSZ, 0);
  // hv: [49152,256] x [256,256], scattered to [node][h][xyz]
  wmma_gemm_lds_kernel<4><<<(3072/8)*(16/4), 256, 0, stream>>>(
      vhA, pWv, nullptr, outHV, 16, 8, DV, 0, 1);
  // es: [491520,64] x [64,32]
  wmma_gemm_lds_kernel<2><<<(30720/8)*(2/2), 256, 0, stream>>>(
      efeat, pWe, edge_ws_b, outES, 2, 2, KE, ESD, 0);

  // 6) layernorms (+ post-LN confidence projection into hs)
  ln_hs_kernel<<<NNODES, 256, 0, stream>>>(outHS, node_ln_g, node_ln_b,
                                           confidence, conf_w, conf_b);
  ln_hv_kernel<<<NNODES, 256, 0, stream>>>(outHV);
  ln_es_kernel<<<NEDGES/256, 256, 0, stream>>>(outES, edge_ln_g, edge_ln_b);
}
